// Transformer_16131897164139
// MI455X (gfx1250) — matrix operand order repaired and checked by reading, not compiled, not run on hardware
//
#include <hip/hip_runtime.h>

// ---------------- model constants ----------------
constexpr int S   = 2048;    // sequence length
constexpr int DM  = 1024;    // d_model
constexpr int HH  = 16;      // heads
constexpr int DH  = 64;      // d_head
constexpr int DFF = 4096;    // d_ff
constexpr int NL  = 12;      // blocks
constexpr int NV  = 50257;   // vocab
constexpr float LN_EPS = 1e-5f;

typedef __bf16 bf16;
typedef __attribute__((ext_vector_type(16))) __bf16 v16bf;
typedef __attribute__((ext_vector_type(8)))  __bf16 v8bf;
typedef __attribute__((ext_vector_type(8)))  float  v8f;
typedef unsigned int v4u __attribute__((ext_vector_type(4)));
typedef int          v8i __attribute__((ext_vector_type(8)));
typedef int          v4i __attribute__((ext_vector_type(4)));

#define DEVFN __device__ __forceinline__

// ---------------- CDNA5 async global->LDS copy (gfx1250) ----------------
DEVFN void async_copy_b128(unsigned int lds_byte_off, const void* gaddr) {
  asm volatile("global_load_async_to_lds_b128 %0, %1, off"
               :
               : "v"(lds_byte_off), "v"((unsigned long long)(uintptr_t)gaddr)
               : "memory");
}
DEVFN void wait_async0() {
  asm volatile("s_wait_asynccnt 0x0" ::: "memory");
}

// ---------------- CDNA5 Tensor Data Mover: contiguous tile -> LDS ----------------
// Builds a D# (groups 0/1 per CDNA5 ISA) for a linear
// run of `nelem` bf16 elements (tile_dim0=nelem, tile_dim1=1) and issues
// TENSOR_LOAD_TO_LDS. Completion tracked with TENSORcnt.
// This toolchain's builtin takes 6 args (g0, g1, g2, g3, g4, cpol).
DEVFN void tdm_load_1d(unsigned int lds_byte_addr, const void* gptr,
                       unsigned int nelem) {
  unsigned long long ga = (unsigned long long)(uintptr_t)gptr;
  v4u g0;
  g0[0] = 1u;                         // count=1, is_restore=0, gather off
  g0[1] = lds_byte_addr;              // lds_addr
  g0[2] = (unsigned int)ga;           // global_addr[31:0]
  g0[3] = (unsigned int)((ga >> 32) & 0x01FFFFFFu) | 0x80000000u;  // addr[56:32] | type=2
  v8i g1;
  g1[0] = 0x00010000;                               // wg_mask=0, data_size=1 (2B)
  g1[1] = (int)((nelem & 0xFFFFu) << 16);           // tensor_dim0 lo16
  g1[2] = (int)(((nelem >> 16) & 0xFFFFu) | (1u << 16)); // dim0 hi | tensor_dim1=1
  g1[3] = (int)((nelem & 0xFFFFu) << 16);           // tile_dim0 = nelem (<=65535)
  g1[4] = 1;                                        // tile_dim1=1, tile_dim2=0
  g1[5] = (int)nelem;                               // tensor_dim0_stride lo32
  g1[6] = 0;                                        // stride hi, dim1_stride lo
  g1[7] = 0;
  v4i z4 = {0, 0, 0, 0};
  v8i z8 = {0, 0, 0, 0, 0, 0, 0, 0};
  __builtin_amdgcn_tensor_load_to_lds(g0, g1, z4, z4, z8, 0);
}
DEVFN void wait_tensor0() { __builtin_amdgcn_s_wait_tensorcnt(0); }

// ---------------- WMMA fragment loaders (CDNA5 wave32 layouts) ----------------
// A-matrix 16x32 bf16, row-major src with leading dim ld.
DEVFN v16bf load_A_bf16(const bf16* base, int ld) {
  int lane = threadIdx.x & 31;
  int m  = lane & 15;
  int kb = (lane & 16) ? 8 : 0;
  v16bf r;
#pragma unroll
  for (int v = 0; v < 8; ++v) {
    int k = (v < 4) ? (kb + 2 * v) : (16 + kb + 2 * (v - 4));
    r[2 * v]     = base[(size_t)m * ld + k];
    r[2 * v + 1] = base[(size_t)m * ld + k + 1];
  }
  return r;
}

// B-matrix 32x16 bf16, row-major src: frag[s] = B[K=table(h,s)][N=lane%16]
DEVFN v16bf load_B_bf16(const bf16* base, long ld) {
  int lane = threadIdx.x & 31;
  int n  = lane & 15;
  int kb = (lane & 16) ? 8 : 0;
  v16bf r;
#pragma unroll
  for (int i = 0; i < 16; ++i) {
    int k = (i < 8) ? (kb + i) : (16 + kb + (i - 8));
    r[i] = base[(long)k * ld + n];
  }
  return r;
}

// Guarded B from fp32 (N-boundary, possibly unaligned), converts to bf16.
DEVFN v16bf load_B_f32_guard(const float* base, long ld, int nvalid) {
  int lane = threadIdx.x & 31;
  int n  = lane & 15;
  int kb = (lane & 16) ? 8 : 0;
  v16bf r;
#pragma unroll
  for (int i = 0; i < 16; ++i) {
    int k = (i < 8) ? (kb + i) : (16 + kb + (i - 8));
    r[i] = (n < nvalid) ? (bf16)base[(long)k * ld + n] : (bf16)0.0f;
  }
  return r;
}

DEVFN v8f wmma_bf16(v16bf a, v16bf b, v8f c) {
  return __builtin_amdgcn_wmma_f32_16x16x32_bf16(false, a, false, b, (short)0, c,
                                                 false, false);
}

DEVFN float wred_sum(float x) {
#pragma unroll
  for (int off = 16; off > 0; off >>= 1) x += __shfl_xor(x, off, 32);
  return x;
}

// ---------------- embed + positional encoding ----------------
__global__ __launch_bounds__(256) void embed_kernel(const int* __restrict__ tok,
                                                    const float* __restrict__ emb,
                                                    float* __restrict__ xo,
                                                    bf16* __restrict__ xob) {
  int s = blockIdx.x;
  int t = threadIdx.x;
  int token = tok[s];
#pragma unroll
  for (int i = 0; i < DM / 256; ++i) {
    int d = t + 256 * i;
    float hi  = (float)(d & ~1);
    float ang = (float)s * __powf(10000.0f, -hi / (float)DM);
    float pe  = (d & 1) ? __cosf(ang) : __sinf(ang);
    float v   = emb[(size_t)token * DM + d] + pe;
    xo[(size_t)s * DM + d]  = v;
    xob[(size_t)s * DM + d] = (bf16)v;
  }
}

// ---------------- LDS-staged double-buffered WMMA GEMM ----------------
// C[M,N] = A_bf16[M,K] * B_f32[K,N]  (+bias)(+resid)(relu)
// Block tile 128x64, K-step 32, 8 waves (4x2), wave tile 32x32 (4 WMMAs/step).
// A staged via async global->LDS b128 (ASYNCcnt); B loaded fp32, cvt, ds_store.
// SMODE: 0 -> f32 to Cf, 1 -> bf16 to Cb (ld=N), 2 -> bf16 transposed Cb[n*M+m]
// BVEC: B rows 16B-aligned (float4 path) vs scalar loads (odd-N weights).
template <bool BIAS, bool RELU, bool RESID, int SMODE, bool BVEC>
__global__ __launch_bounds__(256) void gemm_lds(
    const bf16* __restrict__ A, const float* __restrict__ B,
    const float* __restrict__ bias, const float* __restrict__ resid,
    float* __restrict__ Cf, bf16* __restrict__ Cb,
    int M, int N, int K, long strideBz, long strideCz) {
  constexpr int MB = 128, NB = 64, KS = 32;
  __shared__ __align__(16) bf16 Abuf[2 * MB * KS];  // 16 KB
  __shared__ __align__(16) bf16 Bbuf[2 * KS * NB];  //  8 KB
  int t = threadIdx.x;
  int w = t >> 5, lane = t & 31;
  int wm = w >> 1, wn = w & 1;  // 4 x 2 wave grid
  int m0 = blockIdx.x * MB;
  int n0 = blockIdx.y * NB;
  long z = blockIdx.z;
  B += z * strideBz;
  float* CfL = Cf;
  bf16*  CbL = Cb;
  if (SMODE == 0) { if (CfL) CfL += z * strideCz; }
  else            { CbL += z * strideCz; }

  int arow = w * 16 + (lane >> 2);
  int acg  = (lane & 3) * 8;
  unsigned int lds_a_base =
      (unsigned int)(uintptr_t)(&Abuf[0]) + (unsigned int)(arow * KS + acg) * 2;
  int brow = t >> 3;
  int bcol = (t & 7) * 8;

  v8f acc[2][2] = {};
  float br[8];

  // ---- prologue: stage k-tile 0 into buffer 0 ----
  {
    const float* p = B + (size_t)brow * N + n0 + bcol;
    if (BVEC) {
      float4 a = *(const float4*)p, b = *(const float4*)(p + 4);
      br[0] = a.x; br[1] = a.y; br[2] = a.z; br[3] = a.w;
      br[4] = b.x; br[5] = b.y; br[6] = b.z; br[7] = b.w;
    } else {
#pragma unroll
      for (int i = 0; i < 8; ++i) br[i] = p[i];
    }
    const bf16* g = A + (size_t)(m0 + arow) * K + acg;
    async_copy_b128(lds_a_base, g);
    async_copy_b128(lds_a_base + 8 * KS * 2, g + (size_t)8 * K);
    v8bf pk;
#pragma unroll
    for (int i = 0; i < 8; ++i) pk[i] = (bf16)br[i];
    *(v8bf*)&Bbuf[brow * NB + bcol] = pk;
  }
  wait_async0();
  __syncthreads();

  int buf = 0;
  for (int k0 = 0; k0 < K; k0 += KS) {
    int nk = k0 + KS;
    if (nk < K) {
      const float* p = B + (size_t)(nk + brow) * N + n0 + bcol;
      if (BVEC) {
        float4 a = *(const float4*)p, b = *(const float4*)(p + 4);
        br[0] = a.x; br[1] = a.y; br[2] = a.z; br[3] = a.w;
        br[4] = b.x; br[5] = b.y; br[6] = b.z; br[7] = b.w;
      } else {
#pragma unroll
        for (int i = 0; i < 8; ++i) br[i] = p[i];
      }
      unsigned int l0 = lds_a_base + (unsigned int)((buf ^ 1) * MB * KS) * 2;
      const bf16* g = A + (size_t)(m0 + arow) * K + nk + acg;
      async_copy_b128(l0, g);
      async_copy_b128(l0 + 8 * KS * 2, g + (size_t)8 * K);
    }
    const bf16* Ab = &Abuf[buf * MB * KS];
    const bf16* Bb = &Bbuf[buf * KS * NB];
    v16bf a0 = load_A_bf16(Ab + (wm * 32) * KS, KS);
    v16bf a1 = load_A_bf16(Ab + (wm * 32 + 16) * KS, KS);
    v16bf b0 = load_B_bf16(Bb + wn * 32, NB);
    v16bf b1 = load_B_bf16(Bb + wn * 32 + 16, NB);
    acc[0][0] = wmma_bf16(a0, b0, acc[0][0]);
    acc[0][1] = wmma_bf16(a0, b1, acc[0][1]);
    acc[1][0] = wmma_bf16(a1, b0, acc[1][0]);
    acc[1][1] = wmma_bf16(a1, b1, acc[1][1]);
    if (nk < K) {
      v8bf pk;
#pragma unroll
      for (int i = 0; i < 8; ++i) pk[i] = (bf16)br[i];
      *(v8bf*)&Bbuf[(buf ^ 1) * KS * NB + brow * NB + bcol] = pk;
      wait_async0();
      __syncthreads();
      buf ^= 1;
    }
  }

  // ---- epilogue ----
  int off8 = (lane & 16) ? 8 : 0;
  int nc   = lane & 15;
#pragma unroll
  for (int i = 0; i < 2; ++i) {
#pragma unroll
    for (int j = 0; j < 2; ++j) {
      int mBase = m0 + wm * 32 + i * 16;
      int nn    = n0 + wn * 32 + j * 16 + nc;
      float bval = BIAS ? bias[nn] : 0.0f;
#pragma unroll
      for (int e = 0; e < 8; ++e) {
        int m = mBase + e + off8;
        float v = acc[i][j][e] + bval;
        if (RESID) v += resid[(size_t)m * N + nn];
        if (RELU)  v = fmaxf(v, 0.0f);
        if (SMODE == 0)      CfL[(size_t)m * N + nn] = v;
        else if (SMODE == 1) CbL[(size_t)m * N + nn] = (bf16)v;
        else                 CbL[(size_t)nn * M + m] = (bf16)v;
      }
    }
  }
}

// -------- direct-global WMMA GEMM (ragged N tail of the vocab) --------
template <bool BIAS>
__global__ __launch_bounds__(256) void gemm_ragged(
    const bf16* __restrict__ A, const float* __restrict__ B,
    const float* __restrict__ bias, float* __restrict__ Cf,
    int M, int N, int K, int ncol0) {
  int w    = threadIdx.x >> 5;
  int lane = threadIdx.x & 31;
  int m0 = blockIdx.x * 64 + (w >> 1) * 16;
  int n0 = ncol0 + blockIdx.y * 32 + (w & 1) * 16;
  if (m0 >= M) return;

  v8f acc = {};
  for (int k0 = 0; k0 < K; k0 += 32) {
    v16bf a = load_A_bf16(A + (size_t)m0 * K + k0, K);
    v16bf b = load_B_f32_guard(B + (size_t)k0 * N + n0, N, N - n0);
    acc = wmma_bf16(a, b, acc);
  }

  int off8 = (lane & 16) ? 8 : 0;
  int nc   = lane & 15;
  int n    = n0 + nc;
  if (n >= N) return;
  float bval = BIAS ? bias[n] : 0.0f;
#pragma unroll
  for (int e = 0; e < 8; ++e) {
    int m = m0 + e + off8;
    Cf[(size_t)m * N + n] = acc[e] + bval;
  }
}

// ------- fused causal flash attention (transposed scores, TDM-fed K/V) -------
// One wave per (head, 16-query tile). Qt:[H][64][S], K:[H][S][64], V:[H][S][64]
// K/V 32-key tiles (4 KB each, contiguous) are DMA'd into double-buffered LDS
// by the Tensor Data Mover, prefetched one step ahead; the wave's inner loop
// touches only LDS. Scores computed transposed so softmax stats are per-lane.
__global__ __launch_bounds__(32) void attn_flash(const bf16* __restrict__ Qt,
                                                 const bf16* __restrict__ Kg,
                                                 const bf16* __restrict__ Vg,
                                                 bf16* __restrict__ mha) {
  constexpr float NEG = -1e30f;
  constexpr int TILE = 32 * DH;  // elements per K/V tile
  const float scale = 0.125f;    // 1/sqrt(64)
  int h  = blockIdx.y;
  int i0 = blockIdx.x * 16;
  const bf16* Qth = Qt + (size_t)h * DH * S;  // [64][S]
  const bf16* Kh  = Kg + (size_t)h * S * DH;  // [S][64]
  const bf16* Vh  = Vg + (size_t)h * S * DH;  // [S][64]
  int lane = threadIdx.x;
  int off8 = (lane & 16) ? 8 : 0;
  int q    = lane & 15;
  int qg   = i0 + q;

  __shared__ __align__(16) bf16 Kbuf[2 * TILE];  // 8 KB
  __shared__ __align__(16) bf16 Vbuf[2 * TILE];  // 8 KB
  __shared__ __align__(16) bf16 pt[16 * 32];     // 1 KB
  unsigned int kbase = (unsigned int)(uintptr_t)&Kbuf[0];
  unsigned int vbase = (unsigned int)(uintptr_t)&Vbuf[0];

  // Q^T B-fragments (fixed for the block)
  v16bf qb0 = load_B_bf16(Qth + i0, S);
  v16bf qb1 = load_B_bf16(Qth + (size_t)32 * S + i0, S);

  float mq = NEG, lq = 0.0f;
  v8f acc[4] = {};

  // prologue: DMA first K/V tiles into buffer 0
  tdm_load_1d(kbase, Kh, TILE);
  tdm_load_1d(vbase, Vh, TILE);
  wait_tensor0();
  __syncthreads();

  int buf = 0;
  for (int j0 = 0; j0 < i0 + 16; j0 += 32) {
    int jn = j0 + 32;
    if (jn < i0 + 16) {  // prefetch next tiles into the other buffer
      unsigned int boff = (unsigned int)((buf ^ 1) * TILE) * 2;
      tdm_load_1d(kbase + boff, Kh + (size_t)jn * DH, TILE);
      tdm_load_1d(vbase + boff, Vh + (size_t)jn * DH, TILE);
    }
    const bf16* Kb = &Kbuf[buf * TILE];
    const bf16* Vb = &Vbuf[buf * TILE];

    float t0[8], t1[8];
    {
      v16bf ka0 = load_A_bf16(Kb, DH);
      v16bf ka1 = load_A_bf16(Kb + 32, DH);
      v8f sc = {};
      sc = wmma_bf16(ka0, qb0, sc);
      sc = wmma_bf16(ka1, qb1, sc);
#pragma unroll
      for (int e = 0; e < 8; ++e) {
        int key = j0 + e + off8;
        t0[e] = (key <= qg) ? sc[e] * scale : NEG;
      }
    }
    int j1 = j0 + 16;
    if (j1 < i0 + 16) {
      v16bf ka0 = load_A_bf16(Kb + 16 * DH, DH);
      v16bf ka1 = load_A_bf16(Kb + 16 * DH + 32, DH);
      v8f sc = {};
      sc = wmma_bf16(ka0, qb0, sc);
      sc = wmma_bf16(ka1, qb1, sc);
#pragma unroll
      for (int e = 0; e < 8; ++e) {
        int key = j1 + e + off8;
        t1[e] = (key <= qg) ? sc[e] * scale : NEG;
      }
    } else {
#pragma unroll
      for (int e = 0; e < 8; ++e) t1[e] = NEG;
    }

    // per-query stats: in-lane reduce + single cross-half combine
    float lm = NEG;
#pragma unroll
    for (int e = 0; e < 8; ++e) lm = fmaxf(lm, fmaxf(t0[e], t1[e]));
    lm = fmaxf(lm, __shfl_xor(lm, 16, 32));
    float mnew  = fmaxf(mq, lm);
    float alpha = __expf(mq - mnew);
    float ls = 0.0f;
    v8bf pk0, pk1;
#pragma unroll
    for (int e = 0; e < 8; ++e) {
      float p0 = __expf(t0[e] - mnew);
      float p1 = __expf(t1[e] - mnew);
      ls += p0 + p1;
      pk0[e] = (bf16)p0;
      pk1[e] = (bf16)p1;
    }
    ls += __shfl_xor(ls, 16, 32);
    lq = lq * alpha + ls;
    mq = mnew;

    *(v8bf*)&pt[q * 32 + off8]      = pk0;
    *(v8bf*)&pt[q * 32 + 16 + off8] = pk1;

#pragma unroll
    for (int e = 0; e < 8; ++e) {
      float ae = __shfl(alpha, e + off8, 32);
      acc[0][e] *= ae; acc[1][e] *= ae; acc[2][e] *= ae; acc[3][e] *= ae;
    }
    __syncthreads();
    v16bf pa = load_A_bf16(pt, 32);
#pragma unroll
    for (int c = 0; c < 4; ++c) {
      v16bf vfr = load_B_bf16(Vb + c * 16, DH);
      acc[c] = wmma_bf16(pa, vfr, acc[c]);
    }
    __syncthreads();
    if (jn < i0 + 16) {
      wait_tensor0();
      buf ^= 1;
    }
  }

  float inv = 1.0f / lq;
#pragma unroll
  for (int e = 0; e < 8; ++e) {
    float ie = __shfl(inv, e + off8, 32);
    int srow = i0 + e + off8;
#pragma unroll
    for (int c = 0; c < 4; ++c)
      mha[(size_t)srow * DM + h * DH + c * 16 + q] = (bf16)(acc[c][e] * ie);
  }
}

// ---------------- layernorm (row per block) ----------------
__global__ __launch_bounds__(256) void ln_kernel(const float* __restrict__ y,
                                                 const float* __restrict__ g,
                                                 const float* __restrict__ bb,
                                                 float* __restrict__ xo,
                                                 bf16* __restrict__ xob) {
  int s = blockIdx.x, t = threadIdx.x;
  int lane = t & 31, w = t >> 5;
  const float* row = y + (size_t)s * DM;
  float v[DM / 256];
  float sum = 0.0f;
#pragma unroll
  for (int i = 0; i < DM / 256; ++i) { v[i] = row[t + 256 * i]; sum += v[i]; }
  __shared__ float sm[8];
  __shared__ float stat[2];
  sum = wred_sum(sum);
  if (lane == 0) sm[w] = sum;
  __syncthreads();
  if (t == 0) {
    float tot = 0;
#pragma unroll
    for (int i = 0; i < 8; ++i) tot += sm[i];
    stat[0] = tot * (1.0f / DM);
  }
  __syncthreads();
  float mu = stat[0];
  float d2 = 0.0f;
#pragma unroll
  for (int i = 0; i < DM / 256; ++i) { float d = v[i] - mu; d2 += d * d; }
  d2 = wred_sum(d2);
  if (lane == 0) sm[w] = d2;
  __syncthreads();
  if (t == 0) {
    float tot = 0;
#pragma unroll
    for (int i = 0; i < 8; ++i) tot += sm[i];
    stat[1] = rsqrtf(tot * (1.0f / DM) + LN_EPS);
  }
  __syncthreads();
  float rstd = stat[1];
#pragma unroll
  for (int i = 0; i < DM / 256; ++i) {
    int d = t + 256 * i;
    float nv = (v[i] - mu) * rstd * g[d] + bb[d];
    xo[(size_t)s * DM + d]  = nv;
    xob[(size_t)s * DM + d] = (bf16)nv;
  }
}

// ---------------- host orchestration ----------------
extern "C" void kernel_launch(void* const* d_in, const int* in_sizes, int n_in,
                              void* d_out, int out_size, void* d_ws, size_t ws_size,
                              hipStream_t stream) {
  (void)in_sizes; (void)n_in; (void)out_size; (void)ws_size;
  const int*   tokens = (const int*)  d_in[0];
  const float* emb    = (const float*)d_in[1];
  const float* Wq     = (const float*)d_in[2];
  const float* Wk     = (const float*)d_in[3];
  const float* Wv     = (const float*)d_in[4];
  const float* Wo     = (const float*)d_in[5];
  const float* bo     = (const float*)d_in[6];
  const float* ln1g   = (const float*)d_in[7];
  const float* ln1b   = (const float*)d_in[8];
  const float* W1     = (const float*)d_in[9];
  const float* b1     = (const float*)d_in[10];
  const float* W2     = (const float*)d_in[11];
  const float* b2     = (const float*)d_in[12];
  const float* ln2g   = (const float*)d_in[13];
  const float* ln2b   = (const float*)d_in[14];
  const float* Wout   = (const float*)d_in[15];
  const float* bout   = (const float*)d_in[16];

  char* p = (char*)d_ws;
  float* xf32 = (float*)p; p += (size_t)S * DM * 4;
  float* yf32 = (float*)p; p += (size_t)S * DM * 4;
  bf16*  xb   = (bf16*) p; p += (size_t)S * DM * 2;
  bf16*  qtb  = (bf16*) p; p += (size_t)HH * DH * S * 2;
  bf16*  kb   = (bf16*) p; p += (size_t)HH * S * DH * 2;
  bf16*  vbuf = (bf16*) p; p += (size_t)HH * S * DH * 2;
  bf16*  mhab = (bf16*) p; p += (size_t)S * DM * 2;
  bf16*  h1b  = (bf16*) p; p += (size_t)S * DFF * 2;

  embed_kernel<<<S, 256, 0, stream>>>(tokens, emb, xf32, xb);

  for (int l = 0; l < NL; ++l) {
    const float* Wql = Wq + (size_t)l * HH * DM * DH;
    const float* Wkl = Wk + (size_t)l * HH * DM * DH;
    const float* Wvl = Wv + (size_t)l * HH * DM * DH;
    const float* Wol = Wo + (size_t)l * DM * DM;
    const float* bol = bo + (size_t)l * DM;
    const float* W1l = W1 + (size_t)l * DM * DFF;
    const float* b1l = b1 + (size_t)l * DFF;
    const float* W2l = W2 + (size_t)l * DFF * DM;
    const float* b2l = b2 + (size_t)l * DM;

    dim3 gqkv(S / 128, DH / 64, HH);
    // Q stored transposed: Qt [H][64][S] bf16
    gemm_lds<false, false, false, 2, true><<<gqkv, 256, 0, stream>>>(
        xb, Wql, nullptr, nullptr, nullptr, qtb, S, DH, DM,
        (long)DM * DH, (long)DH * S);
    // K [H][S][64] bf16
    gemm_lds<false, false, false, 1, true><<<gqkv, 256, 0, stream>>>(
        xb, Wkl, nullptr, nullptr, nullptr, kb, S, DH, DM,
        (long)DM * DH, (long)S * DH);
    // V [H][S][64] bf16
    gemm_lds<false, false, false, 1, true><<<gqkv, 256, 0, stream>>>(
        xb, Wvl, nullptr, nullptr, nullptr, vbuf, S, DH, DM,
        (long)DM * DH, (long)S * DH);

    attn_flash<<<dim3(S / 16, HH), 32, 0, stream>>>(qtb, kb, vbuf, mhab);

    // out-proj + bias + residual -> y (f32)
    gemm_lds<true, false, true, 0, true><<<dim3(S / 128, DM / 64, 1), 256, 0, stream>>>(
        mhab, Wol, bol, xf32, yf32, nullptr, S, DM, DM, 0, 0);
    ln_kernel<<<S, 256, 0, stream>>>(yf32, ln1g + (size_t)l * DM,
                                     ln1b + (size_t)l * DM, xf32, xb);

    // FFN up + relu -> h1 (bf16)
    gemm_lds<true, true, false, 1, true><<<dim3(S / 128, DFF / 64, 1), 256, 0, stream>>>(
        xb, W1l, b1l, nullptr, nullptr, h1b, S, DFF, DM, 0, 0);
    // FFN down + bias + residual -> y (f32)
    gemm_lds<true, false, true, 0, true><<<dim3(S / 128, DM / 64, 1), 256, 0, stream>>>(
        h1b, W2l, b2l, xf32, yf32, nullptr, S, DM, DFF, 0, 0);
    ln_kernel<<<S, 256, 0, stream>>>(yf32, ln2g + (size_t)l * DM,
                                     ln2b + (size_t)l * DM, xf32, xb);
  }

  // logits: bulk columns via LDS kernel (scalar B loads: N is odd), tail ragged
  constexpr int NVMAIN = (NV / 64) * 64;  // 50240
  gemm_lds<true, false, false, 0, false><<<dim3(S / 128, NVMAIN / 64, 1), 256, 0, stream>>>(
      xb, Wout, bout, nullptr, (float*)d_out, nullptr, S, NV, DM, 0, 0);
  gemm_ragged<true><<<dim3(S / 64, 1, 1), 256, 0, stream>>>(
      xb, Wout, bout, (float*)d_out, S, NV, DM, NVMAIN);
}
